// Autoformer_14044543058172
// MI455X (gfx1250) — matrix operand order repaired and checked by reading, not compiled, not run on hardware
//
#include <hip/hip_runtime.h>
#include <hip/hip_bf16.h>
#include <math.h>

namespace {

constexpr int Bn    = 128;
constexpr int Ls    = 192;    // SEQ_LEN == DEC_LEN
constexpr int Dm    = 512;
constexpr int Dff   = 2048;
constexpr int Cin   = 6;
constexpr int Nmark = 5;
constexpr int PRED  = 96;
constexpr int LBL   = 96;
constexpr int KPAD  = 12;     // (25-1)/2
constexpr int TOPK  = 5;      // int(1*log(192))
constexpr int TOK   = Bn * Ls;

typedef _Float16 h16;
typedef __attribute__((ext_vector_type(16))) _Float16 v16h;
typedef __attribute__((ext_vector_type(8)))  _Float16 v8h;
typedef __attribute__((ext_vector_type(8)))  float    v8f;

// ---------------------------------------------------------------------------
// WMMA GEMM:  C[M,N] = act(A[M,K] @ B[K,N] + bias)
//   A f16 row-major [M,K]; Bt f16 = B transposed [N,K] row-major.
// Block = 256 threads = 8 waves (4M x 2N), block tile 128(M) x 64(N).
// Wave tile 32x32 -> 4 accumulators, 4 v_wmma per K-step.
// The 64-wide B panel is staged in LDS (double buffered, 8KB) to kill the
// 4x intra-block refetch; A streams straight from global/L2.
// ---------------------------------------------------------------------------
__global__ __launch_bounds__(256)
void k_gemm(const h16* __restrict__ A, const h16* __restrict__ Bt,
            const float* __restrict__ bias,
            float* __restrict__ C, h16* __restrict__ Ch,
            int M, int N, int K, int gelu)
{
    __shared__ h16 smB[2][64][32];

    const int tid  = threadIdx.x;
    const int lane = tid & 31;
    const int wave = tid >> 5;
    const int mTile  = blockIdx.y * 128 + (wave & 3) * 32;
    const int nTile  = blockIdx.x * 64 + (wave >> 2) * 32;
    const int nBlock = blockIdx.x * 64;

    const int row  = lane & 15;
    const int aSel = (lane >> 4) << 3;   // A: halves [0..7]=kb+0..7, [8..15]=kb+16..23
    const int bSel = (lane >> 4) << 3;   // B: halves [0..7]=kb+0..7, [8..15]=kb+16..23

    const h16* aP0 = A + (long)(mTile + row) * K + aSel;
    const h16* aP1 = aP0 + 16L * K;

    // cooperative B fill: one b128 per thread per K-step (256*16B = 4KB tile)
    const int rowF = tid >> 2;            // 0..63 (local n)
    const int chkF = (tid & 3) << 3;      // 0,8,16,24 (half offset in K window)
    const h16* bF  = Bt + (long)(nBlock + rowF) * K + chkF;

    const int nL0 = (wave >> 2) * 32 + (lane & 15);
    const int nL1 = nL0 + 16;

    v8f acc00 = {}, acc01 = {}, acc10 = {}, acc11 = {};

    // prologue fill of buffer 0 (K-window [0,32))
    *(v8h*)&smB[0][rowF][chkF] = *(const v8h*)(bF);

    int it = 0;
    for (int k0 = 0; k0 < K; k0 += 32, ++it) {
        const int p = it & 1;
        __syncthreads();                      // buf p ready; prior reads of p^1 done
        if (k0 + 32 < K)
            *(v8h*)&smB[p ^ 1][rowF][chkF] = *(const v8h*)(bF + k0 + 32);

        union { v16h v; v8h h[2]; } a0, a1, b0, b1;
        a0.h[0] = *(const v8h*)(aP0 + k0);
        a0.h[1] = *(const v8h*)(aP0 + k0 + 16);
        a1.h[0] = *(const v8h*)(aP1 + k0);
        a1.h[1] = *(const v8h*)(aP1 + k0 + 16);
        b0.h[0] = *(const v8h*)&smB[p][nL0][bSel];
        b0.h[1] = *(const v8h*)&smB[p][nL0][bSel + 16];
        b1.h[0] = *(const v8h*)&smB[p][nL1][bSel];
        b1.h[1] = *(const v8h*)&smB[p][nL1][bSel + 16];

        acc00 = __builtin_amdgcn_wmma_f32_16x16x32_f16(false, a0.v, false, b0.v,
                                                       (short)0, acc00, false, false);
        acc01 = __builtin_amdgcn_wmma_f32_16x16x32_f16(false, a0.v, false, b1.v,
                                                       (short)0, acc01, false, false);
        acc10 = __builtin_amdgcn_wmma_f32_16x16x32_f16(false, a1.v, false, b0.v,
                                                       (short)0, acc10, false, false);
        acc11 = __builtin_amdgcn_wmma_f32_16x16x32_f16(false, a1.v, false, b1.v,
                                                       (short)0, acc11, false, false);
    }

    const int col  = lane & 15;
    const int rOff = (lane >> 4) << 3;
    auto store = [&](const v8f& acc, int mBase, int nBase) {
        for (int r = 0; r < 8; ++r) {
            int m = mBase + rOff + r;
            if (m >= M) continue;
            int n = nBase + col;
            if (n >= N) continue;
            float v = acc[r];
            if (bias) v += bias[n];
            if (gelu) v = 0.5f * v * (1.f + erff(v * 0.70710678118f));
            long i0 = (long)m * N + n;
            if (C)  C[i0]  = v;
            if (Ch) Ch[i0] = (h16)v;
        }
    };
    store(acc00, mTile,      nTile);
    store(acc01, mTile,      nTile + 16);
    store(acc10, mTile + 16, nTile);
    store(acc11, mTile + 16, nTile + 16);
}

__global__ void k_cvt_h(const float* __restrict__ in, h16* __restrict__ out, long n) {
    long i = (long)blockIdx.x * blockDim.x + threadIdx.x;
    if (i < n) out[i] = (h16)in[i];
}

__global__ void k_trans_h(const float* __restrict__ W, h16* __restrict__ Wt, int K, int N) {
    long i = (long)blockIdx.x * blockDim.x + threadIdx.x;
    if (i >= (long)K * N) return;
    int n = (int)(i % N); int k = (int)(i / N);
    Wt[(long)n * K + k] = (h16)W[i];
}

__global__ void k_add(const float* __restrict__ a, const float* __restrict__ b,
                      float* __restrict__ o, long n) {
    long i = (long)blockIdx.x * blockDim.x + threadIdx.x;
    if (i < n) o[i] = a[i] + b[i];
}

// series_decomp: moving average k=25 with edge-replicate padding (== clamp)
__global__ void k_decomp(const float* __restrict__ x, float* __restrict__ seas,
                         float* __restrict__ mean, int L, int D, int accMean, long n)
{
    long i = (long)blockIdx.x * blockDim.x + threadIdx.x;
    if (i >= n) return;
    int d = (int)(i % D); long r = i / D; int t = (int)(r % L); long b = r / L;
    const float* xb = x + b * (long)L * D + d;
    float s = 0.f;
    for (int j = -KPAD; j <= KPAD; ++j) {
        int tt = t + j; tt = tt < 0 ? 0 : (tt >= L ? L - 1 : tt);
        s += xb[(long)tt * D];
    }
    s *= (1.f / 25.f);
    if (seas) seas[i] = x[i] - s;
    if (mean) { if (accMean) mean[i] += s; else mean[i] = s; }
}

// embed: circular conv1d (tap=3) over Cin channels + mark @ Wt
__global__ void k_embed(const float* __restrict__ xin, const float* __restrict__ mark,
                        const float* __restrict__ Wc, const float* __restrict__ Wt,
                        float* __restrict__ out, int L)
{
    long n = (long)Bn * L * Dm;
    long i = (long)blockIdx.x * blockDim.x + threadIdx.x;
    if (i >= n) return;
    int d = (int)(i % Dm); long r = i / Dm; int t = (int)(r % L); int b = (int)(r / L);
    float s = 0.f;
    for (int tap = 0; tap < 3; ++tap) {
        int tt = t + tap - 1; tt = (tt + L) % L;
        const float* xr = xin + ((long)b * L + tt) * Cin;
        const float* wr = Wc + (long)tap * Cin * Dm + d;
        for (int c = 0; c < Cin; ++c) s += xr[c] * wr[(long)c * Dm];
    }
    const float* mr = mark + ((long)b * L + t) * Nmark;
    for (int m = 0; m < Nmark; ++m) s += mr[m] * Wt[(long)m * Dm + d];
    out[i] = s;
}

// circular cross-correlation per channel: attn[b][l][d] = sum_t q[b,t,d]*k[b,(t+l)%L,d]
__global__ void k_corr(const float* __restrict__ q, const float* __restrict__ k,
                       float* __restrict__ attn, int L)
{
    long n = (long)Bn * L * Dm;
    long i = (long)blockIdx.x * blockDim.x + threadIdx.x;
    if (i >= n) return;
    int d = (int)(i % Dm); long r = i / Dm; int l = (int)(r % L); int b = (int)(r / L);
    const float* qb = q + (long)b * L * Dm + d;
    const float* kb = k + (long)b * L * Dm + d;
    float s = 0.f;
    for (int t = 0; t < L; ++t) {
        int tl = t + l; if (tl >= L) tl -= L;
        s += qb[(long)t * Dm] * kb[(long)tl * Dm];
    }
    attn[i] = s;
}

__global__ void k_corr_mean(const float* __restrict__ attn, float* __restrict__ mv, int L)
{
    int i = blockIdx.x * blockDim.x + threadIdx.x;
    if (i >= Bn * L) return;
    const float* row = attn + (long)i * Dm;
    float s = 0.f;
    for (int d = 0; d < Dm; ++d) s += row[d];
    mv[i] = s * (1.f / Dm);
}

__global__ void k_topk(const float* __restrict__ mv, int L,
                       float* __restrict__ w, int* __restrict__ dl)
{
    int b = blockIdx.x * blockDim.x + threadIdx.x;
    if (b >= Bn) return;
    int   chosen[TOPK];
    float cw[TOPK];
    for (int kk = 0; kk < TOPK; ++kk) {
        float best = -3.4e38f; int bi = 0;
        for (int l = 0; l < L; ++l) {
            bool skip = false;
            for (int j = 0; j < kk; ++j) if (chosen[j] == l) skip = true;
            float v = mv[b * L + l];
            if (!skip && v > best) { best = v; bi = l; }
        }
        chosen[kk] = bi; cw[kk] = best;
    }
    float mx = cw[0];
    for (int j = 1; j < TOPK; ++j) mx = fmaxf(mx, cw[j]);
    float ssum = 0.f;
    for (int j = 0; j < TOPK; ++j) { cw[j] = expf(cw[j] - mx); ssum += cw[j]; }
    for (int j = 0; j < TOPK; ++j) { w[b * TOPK + j] = cw[j] / ssum; dl[b * TOPK + j] = chosen[j]; }
}

// agg[b,t,d] = sum_k w_k * v[b,(t+delay_k)%L,d]  (written f16 for the Wo GEMM)
__global__ void k_agg(const float* __restrict__ v, const float* __restrict__ w,
                      const int* __restrict__ dl, h16* __restrict__ outh, int L)
{
    long n = (long)Bn * L * Dm;
    long i = (long)blockIdx.x * blockDim.x + threadIdx.x;
    if (i >= n) return;
    int d = (int)(i % Dm); long r = i / Dm; int t = (int)(r % L); int b = (int)(r / L);
    float s = 0.f;
    for (int kk = 0; kk < TOPK; ++kk) {
        int tt = t + dl[b * TOPK + kk]; if (tt >= L) tt -= L;
        s += w[b * TOPK + kk] * v[((long)b * L + tt) * Dm + d];
    }
    outh[i] = (h16)s;
}

// my_layernorm stage 1: per-token affine LN (one wave per token)
__global__ __launch_bounds__(256)
void k_ln1(const float* __restrict__ x, const float* __restrict__ g,
           const float* __restrict__ bb, float* __restrict__ y, int L)
{
    int lane = threadIdx.x & 31, wv = threadIdx.x >> 5;
    long tok = (long)blockIdx.x * 8 + wv;
    if (tok >= (long)Bn * L) return;
    const float* xr = x + tok * Dm;
    float s = 0.f, s2 = 0.f;
    for (int j = lane; j < Dm; j += 32) { float v = xr[j]; s += v; s2 += v * v; }
    for (int m = 16; m >= 1; m >>= 1) { s += __shfl_xor(s, m, 32); s2 += __shfl_xor(s2, m, 32); }
    float mu = s * (1.f / Dm);
    float var = s2 * (1.f / Dm) - mu * mu;
    float inv = rsqrtf(var + 1e-5f);
    float* yr = y + tok * Dm;
    for (int j = lane; j < Dm; j += 32) yr[j] = (xr[j] - mu) * inv * g[j] + bb[j];
}

// my_layernorm stage 2: subtract per-(b,d) mean over time
__global__ void k_ln2(float* __restrict__ y, int L)
{
    int i = blockIdx.x * blockDim.x + threadIdx.x;
    if (i >= Bn * Dm) return;
    int d = i % Dm, b = i / Dm;
    float s = 0.f;
    for (int t = 0; t < L; ++t) s += y[((long)b * L + t) * Dm + d];
    s *= (1.f / L);
    for (int t = 0; t < L; ++t) y[((long)b * L + t) * Dm + d] -= s;
}

// seasonal_init / trend_init for the decoder
__global__ void k_dec_init(const float* __restrict__ x_enc, const float* __restrict__ seas,
                           const float* __restrict__ trend, float* __restrict__ sinit,
                           float* __restrict__ tinit)
{
    int i = blockIdx.x * blockDim.x + threadIdx.x;
    if (i >= Bn * Ls * Cin) return;
    int c = i % Cin; int r = i / Cin; int t = r % Ls; int b = r / Ls;
    if (t < LBL) {
        sinit[i] = seas[((long)b * Ls + (Ls - LBL) + t) * Cin + c];
        tinit[i] = trend[((long)b * Ls + (Ls - LBL) + t) * Cin + c];
    } else {
        sinit[i] = 0.f;
        float s = 0.f;
        for (int tt = 0; tt < Ls; ++tt) s += x_enc[((long)b * Ls + tt) * Cin + c];
        tinit[i] = s * (1.f / Ls);
    }
}

// residual_trend = circ_conv1d(trend_sum[B,L,Dm], Wt[3,Dm,Cin]) accumulated
__global__ void k_trendconv(const float* __restrict__ tsum, const float* __restrict__ Wt,
                            float* __restrict__ tfinal, int L)
{
    int i = blockIdx.x * blockDim.x + threadIdx.x;
    if (i >= Bn * L * Cin) return;
    int c = i % Cin; int r = i / Cin; int t = r % L; int b = r / L;
    float s = 0.f;
    for (int tap = 0; tap < 3; ++tap) {
        int tt = (t + tap - 1 + L) % L;
        const float* xr = tsum + ((long)b * L + tt) * Dm;
        const float* wr = Wt + (long)tap * Dm * Cin + c;
        for (int d = 0; d < Dm; ++d) s += xr[d] * wr[(long)d * Cin];
    }
    tfinal[i] += s;
}

// dec_out = (x @ Wp + bp) + trend, last PRED steps
__global__ void k_final(const float* __restrict__ x, const float* __restrict__ Wp,
                        const float* __restrict__ bp, const float* __restrict__ tfinal,
                        float* __restrict__ out)
{
    int i = blockIdx.x * blockDim.x + threadIdx.x;
    if (i >= Bn * PRED * Cin) return;
    int c = i % Cin; int r = i / Cin; int t = r % PRED; int b = r / PRED;
    int tt = (Ls - PRED) + t;
    const float* xr = x + ((long)b * Ls + tt) * Dm;
    float s = bp[c];
    for (int d = 0; d < Dm; ++d) s += xr[d] * Wp[(long)d * Cin + c];
    out[i] = s + tfinal[((long)b * Ls + tt) * Cin + c];
}

} // namespace

// ---------------------------------------------------------------------------
extern "C" void kernel_launch(void* const* d_in, const int* in_sizes, int n_in,
                              void* d_out, int out_size, void* d_ws, size_t ws_size,
                              hipStream_t stream)
{
    (void)in_sizes; (void)n_in; (void)out_size; (void)ws_size;

    const float* x_enc      = (const float*)d_in[0];
    const float* x_mark_enc = (const float*)d_in[1];
    const float* x_mark_dec = (const float*)d_in[3];   // x_dec values unused by model

    // params flattened (jax tree order: sorted dict keys, depth first)
    int idx = 4;
    auto nx = [&]() { return (const float*)d_in[idx++]; };
    const float *Wp = nx(), *bp = nx();
    const float *dWc = nx(), *dWtm = nx();
    const float *dW1 = nx(), *dW2 = nx(), *dWt = nx();
    const float *cWk = nx(), *cWo = nx(), *cWq = nx(), *cWv = nx();
    const float *cbk = nx(), *cbo = nx(), *cbq = nx(), *cbv = nx();
    const float *sWk = nx(), *sWo = nx(), *sWq = nx(), *sWv = nx();
    const float *sbk = nx(), *sbo = nx(), *sbq = nx(), *sbv = nx();
    const float *dnb = nx(), *dng = nx();
    const float *eWc = nx(), *eWtm = nx();
    struct EncL { const float *W1, *W2, *Wk, *Wo, *Wq, *Wv, *bk, *bo, *bq, *bv; } enc[2];
    for (int l = 0; l < 2; ++l) {
        enc[l].W1 = nx(); enc[l].W2 = nx();
        enc[l].Wk = nx(); enc[l].Wo = nx(); enc[l].Wq = nx(); enc[l].Wv = nx();
        enc[l].bk = nx(); enc[l].bo = nx(); enc[l].bq = nx(); enc[l].bv = nx();
    }
    const float *enb = nx(), *eng = nx();

    // ---- workspace bump allocator -----------------------------------------
    char* cur = (char*)d_ws;
    auto alloc = [&](size_t bytes) -> void* {
        void* p = (void*)cur; cur += (bytes + 255) & ~(size_t)255; return p;
    };
    auto g1 = [](long n) { return dim3((unsigned)((n + 255) / 256)); };

    // weight conversion to f16, transposed [N,K]
    auto mkT = [&](const float* W, int K, int N) -> h16* {
        h16* p = (h16*)alloc((size_t)K * N * sizeof(h16));
        k_trans_h<<<g1((long)K * N), 256, 0, stream>>>(W, p, K, N);
        return p;
    };
    h16* eWqT[2]; h16* eWkT[2]; h16* eWvT[2]; h16* eWoT[2]; h16* eW1T[2]; h16* eW2T[2];
    for (int l = 0; l < 2; ++l) {
        eWqT[l] = mkT(enc[l].Wq, Dm, Dm); eWkT[l] = mkT(enc[l].Wk, Dm, Dm);
        eWvT[l] = mkT(enc[l].Wv, Dm, Dm); eWoT[l] = mkT(enc[l].Wo, Dm, Dm);
        eW1T[l] = mkT(enc[l].W1, Dm, Dff); eW2T[l] = mkT(enc[l].W2, Dff, Dm);
    }
    h16* sWqT = mkT(sWq, Dm, Dm); h16* sWkT = mkT(sWk, Dm, Dm);
    h16* sWvT = mkT(sWv, Dm, Dm); h16* sWoT = mkT(sWo, Dm, Dm);
    h16* cWqT = mkT(cWq, Dm, Dm); h16* cWkT = mkT(cWk, Dm, Dm);
    h16* cWvT = mkT(cWv, Dm, Dm); h16* cWoT = mkT(cWo, Dm, Dm);
    h16* dW1T = mkT(dW1, Dm, Dff); h16* dW2T = mkT(dW2, Dff, Dm);

    const size_t actB = (size_t)TOK * Dm * sizeof(float);
    float* X     = (float*)alloc(actB);
    float* Y     = (float*)alloc(actB);
    float* Qb    = (float*)alloc(actB);
    float* Kb    = (float*)alloc(actB);
    float* Vb    = (float*)alloc(actB);
    float* TS    = (float*)alloc(actB);                 // trend accumulator (decoder)
    float* CORRD = (float*)alloc(actB);                 // decoder corr scratch
    h16*   XH    = (h16*)alloc((size_t)TOK * Dm * sizeof(h16));
    h16*   ENCH  = (h16*)alloc((size_t)TOK * Dm * sizeof(h16));
    h16*   AGG   = (h16*)alloc((size_t)TOK * Dm * sizeof(h16));
    h16*   HID   = (h16*)alloc((size_t)TOK * Dff * sizeof(h16));
    float* MV    = (float*)alloc((size_t)Bn * Ls * sizeof(float));
    float* WT5   = (float*)alloc((size_t)Bn * TOPK * sizeof(float));
    int*   DL5   = (int*)alloc((size_t)Bn * TOPK * sizeof(int));
    float* seas6 = (float*)alloc((size_t)Bn * Ls * Cin * sizeof(float));
    float* trnd6 = (float*)alloc((size_t)Bn * Ls * Cin * sizeof(float));
    float* sinit = (float*)alloc((size_t)Bn * Ls * Cin * sizeof(float));
    float* tinit = (float*)alloc((size_t)Bn * Ls * Cin * sizeof(float));

    const long nAct  = (long)TOK * Dm;
    const dim3 gDD(Dm / 64, TOK / 128);     // GEMM grid, N=512  (block 128x64)
    const dim3 gDF(Dff / 64, TOK / 128);    // GEMM grid, N=2048

    float* outDec = (float*)d_out;                              // [B,96,6]
    float* attn0  = outDec + (size_t)Bn * PRED * Cin;           // [B,L,H,E] layer 0
    float* attn1  = attn0 + (size_t)Bn * Ls * Dm;               // layer 1

    // autocorrelation attention block
    auto ac = [&](const h16* xq16, const h16* xkv16,
                  const h16* WqT, const float* bq, const h16* WkT, const float* bk,
                  const h16* WvT, const float* bv, const h16* WoT, const float* bo,
                  float* attnDst, float* outDst)
    {
        k_gemm<<<gDD, 256, 0, stream>>>(xq16,  WqT, bq, Qb, nullptr, TOK, Dm, Dm, 0);
        k_gemm<<<gDD, 256, 0, stream>>>(xkv16, WkT, bk, Kb, nullptr, TOK, Dm, Dm, 0);
        k_gemm<<<gDD, 256, 0, stream>>>(xkv16, WvT, bv, Vb, nullptr, TOK, Dm, Dm, 0);
        k_corr<<<g1(nAct), 256, 0, stream>>>(Qb, Kb, attnDst, Ls);
        k_corr_mean<<<g1(Bn * Ls), 256, 0, stream>>>(attnDst, MV, Ls);
        k_topk<<<1, 128, 0, stream>>>(MV, Ls, WT5, DL5);
        k_agg<<<g1(nAct), 256, 0, stream>>>(Vb, WT5, DL5, AGG, Ls);
        k_gemm<<<gDD, 256, 0, stream>>>(AGG, WoT, bo, outDst, nullptr, TOK, Dm, Dm, 0);
    };

    // ---------------- encoder ----------------
    k_embed<<<g1(nAct), 256, 0, stream>>>(x_enc, x_mark_enc, eWc, eWtm, X, Ls);
    float* attnDst[2] = { attn0, attn1 };
    for (int l = 0; l < 2; ++l) {
        k_cvt_h<<<g1(nAct), 256, 0, stream>>>(X, XH, nAct);
        ac(XH, XH, eWqT[l], enc[l].bq, eWkT[l], enc[l].bk, eWvT[l], enc[l].bv,
           eWoT[l], enc[l].bo, attnDst[l], Y);
        k_add<<<g1(nAct), 256, 0, stream>>>(X, Y, X, nAct);
        k_decomp<<<g1(nAct), 256, 0, stream>>>(X, Y, nullptr, Ls, Dm, 0, nAct);
        k_cvt_h<<<g1(nAct), 256, 0, stream>>>(Y, XH, nAct);
        k_gemm<<<gDF, 256, 0, stream>>>(XH, eW1T[l], nullptr, nullptr, HID, TOK, Dff, Dm, 1);
        k_gemm<<<gDD, 256, 0, stream>>>(HID, eW2T[l], nullptr, Qb, nullptr, TOK, Dm, Dff, 0);
        k_add<<<g1(nAct), 256, 0, stream>>>(Y, Qb, Qb, nAct);
        k_decomp<<<g1(nAct), 256, 0, stream>>>(Qb, X, nullptr, Ls, Dm, 0, nAct);
    }
    k_ln1<<<dim3((Bn * Ls + 7) / 8), 256, 0, stream>>>(X, eng, enb, Y, Ls);
    k_ln2<<<g1(Bn * Dm), 256, 0, stream>>>(Y, Ls);
    k_cvt_h<<<g1(nAct), 256, 0, stream>>>(Y, ENCH, nAct);   // enc_out (f16 for cross-attn)

    // ---------------- decoder ----------------
    const long n6 = (long)Bn * Ls * Cin;
    k_decomp<<<g1(n6), 256, 0, stream>>>(x_enc, seas6, trnd6, Ls, Cin, 0, n6);
    k_dec_init<<<g1(n6), 256, 0, stream>>>(x_enc, seas6, trnd6, sinit, tinit);
    k_embed<<<g1(nAct), 256, 0, stream>>>(sinit, x_mark_dec, dWc, dWtm, X, Ls);

    // self attention
    k_cvt_h<<<g1(nAct), 256, 0, stream>>>(X, XH, nAct);
    ac(XH, XH, sWqT, sbq, sWkT, sbk, sWvT, sbv, sWoT, sbo, CORRD, Y);
    k_add<<<g1(nAct), 256, 0, stream>>>(X, Y, X, nAct);
    k_decomp<<<g1(nAct), 256, 0, stream>>>(X, Y, TS, Ls, Dm, 0, nAct);   // trend1
    // cross attention (K/V from encoder output)
    k_cvt_h<<<g1(nAct), 256, 0, stream>>>(Y, XH, nAct);
    ac(XH, ENCH, cWqT, cbq, cWkT, cbk, cWvT, cbv, cWoT, cbo, CORRD, X);
    k_add<<<g1(nAct), 256, 0, stream>>>(Y, X, X, nAct);
    k_decomp<<<g1(nAct), 256, 0, stream>>>(X, Y, TS, Ls, Dm, 1, nAct);   // trend2
    // FFN
    k_cvt_h<<<g1(nAct), 256, 0, stream>>>(Y, XH, nAct);
    k_gemm<<<gDF, 256, 0, stream>>>(XH, dW1T, nullptr, nullptr, HID, TOK, Dff, Dm, 1);
    k_gemm<<<gDD, 256, 0, stream>>>(HID, dW2T, nullptr, Qb, nullptr, TOK, Dm, Dff, 0);
    k_add<<<g1(nAct), 256, 0, stream>>>(Y, Qb, Qb, nAct);
    k_decomp<<<g1(nAct), 256, 0, stream>>>(Qb, X, TS, Ls, Dm, 1, nAct);  // trend3
    k_trendconv<<<g1(n6), 256, 0, stream>>>(TS, dWt, tinit, Ls);

    k_ln1<<<dim3((Bn * Ls + 7) / 8), 256, 0, stream>>>(X, dng, dnb, Y, Ls);
    k_ln2<<<g1(Bn * Dm), 256, 0, stream>>>(Y, Ls);
    k_final<<<g1((long)Bn * PRED * Cin), 256, 0, stream>>>(Y, Wp, bp, tinit, outDec);
}
